// NoiseConditionalProteinMPNN_11751030522734
// MI455X (gfx1250) — matrix operand order repaired and checked by reading, not compiled, not run on hardware
//
#include <hip/hip_runtime.h>
#include <math.h>

// ---------------- model constants ----------------
#define Hdim   128
#define Ldim   1024
#define Bdim   8
#define Kneib  32
#define NNODE  (Bdim*Ldim)          // 8192
#define NEDGE  (NNODE*Kneib)        // 262144
#define TCDdim 512
#define EPAD   192                  // 167 edge-feature dims padded to 192 (mult of 32)
#define EFDIM  167

typedef __attribute__((ext_vector_type(16))) __bf16 v16bf;
typedef __attribute__((ext_vector_type(8)))  __bf16 v8bf;
typedef __attribute__((ext_vector_type(8)))  float  v8f;

// ---------------- bf16 helpers ----------------
// native hardware convert (v_cvt_*bf16*) instead of bit-twiddled RNE
__device__ inline __bf16 f2bf(float x) { return (__bf16)x; }
__device__ inline unsigned short f2bfu(float x) {
  union { __bf16 b; unsigned short s; } c; c.b = (__bf16)x; return c.s;
}
__device__ __host__ inline float bfu2f(unsigned short h) {
  union { unsigned u; float f; } c; c.u = ((unsigned)h) << 16;
  return c.f;
}

__device__ inline float gelu_exact(float x) {
  return 0.5f * x * (1.0f + erff(x * 0.70710678118654752f));
}

// block-of-128 LayerNorm helper (each thread owns one channel h)
__device__ inline float block_ln128(float x, int h, const float* g, const float* b,
                                    float* s1, float* s2) {
  s1[h] = x; s2[h] = x * x; __syncthreads();
  for (int s = 64; s > 0; s >>= 1) {
    if (h < s) { s1[h] += s1[h + s]; s2[h] += s2[h + s]; }
    __syncthreads();
  }
  float mean = s1[0] * (1.0f / 128.0f);
  float var  = s2[0] * (1.0f / 128.0f) - mean * mean;
  float rs   = rsqrtf(var + 1e-5f);
  float y = (x - mean) * rs * g[h] + b[h];
  __syncthreads();
  return y;
}

// ---------------- feature kernels ----------------
__global__ void k_extract_ca(const float* __restrict__ coords, float* __restrict__ ca) {
  int i = blockIdx.x * blockDim.x + threadIdx.x;
  if (i < NNODE * 3) {
    int n = i / 3, c = i % 3;
    ca[i] = coords[((long)n * 37 + 1) * 3 + c];   // Ca = coords[:,:,1,:]
  }
}

__global__ void k_knn(const float* __restrict__ ca, const float* __restrict__ mask,
                      int* __restrict__ eidx, float* __restrict__ dn) {
  int n = blockIdx.x * blockDim.x + threadIdx.x;
  if (n >= NNODE) return;
  int b = n >> 10;
  const float* cb = ca + (long)b * Ldim * 3;
  const float* mb = mask + b * Ldim;
  int i = n & (Ldim - 1);
  float xi = cb[i*3], yi = cb[i*3+1], zi = cb[i*3+2];
  float mi = mb[i];
  float rowmax = 0.f;
  for (int j = 0; j < Ldim; ++j) {
    float dx = xi - cb[j*3], dy = yi - cb[j*3+1], dz = zi - cb[j*3+2];
    float d = sqrtf(dx*dx + dy*dy + dz*dz + 1e-6f);
    rowmax = fmaxf(rowmax, mi * mb[j] * d);
  }
  float best[Kneib]; int bidx[Kneib];
  for (int k = 0; k < Kneib; ++k) { best[k] = 3.4e38f; bidx[k] = 0; }
  for (int j = 0; j < Ldim; ++j) {
    float dx = xi - cb[j*3], dy = yi - cb[j*3+1], dz = zi - cb[j*3+2];
    float d = sqrtf(dx*dx + dy*dy + dz*dz + 1e-6f);
    float m2 = mi * mb[j];
    float dadj = m2 * d + (1.f - m2) * rowmax;
    if (dadj < best[Kneib-1]) {
      int p = Kneib - 1;
      while (p > 0 && best[p-1] > dadj) { best[p] = best[p-1]; bidx[p] = bidx[p-1]; --p; }
      best[p] = dadj; bidx[p] = j;
    }
  }
  for (int k = 0; k < Kneib; ++k) { eidx[n*Kneib + k] = bidx[k]; dn[n*Kneib + k] = best[k]; }
}

__device__ inline void norm3(float& x, float& y, float& z) {
  float n = sqrtf(x*x + y*y + z*z);
  float inv = 1.f / fmaxf(n, 1e-12f);
  x *= inv; y *= inv; z *= inv;
}
__device__ inline void cross3(const float* a, const float* b, float* c) {
  c[0] = a[1]*b[2] - a[2]*b[1];
  c[1] = a[2]*b[0] - a[0]*b[2];
  c[2] = a[0]*b[1] - a[1]*b[0];
}

__global__ void k_orient(const float* __restrict__ ca, float* __restrict__ O,
                         float* __restrict__ AD) {
  int n = blockIdx.x * blockDim.x + threadIdx.x;
  if (n >= NNODE) return;
  int b = n >> 10, l = n & (Ldim - 1);
  float Ov[9] = {0,0,0,0,0,0,0,0,0};
  float ADv[3] = {0,0,0};
  if (l >= 1 && l <= Ldim - 3) {
    const float* cb = ca + (long)b * Ldim * 3;
    float u2[3], u1[3], u0[3];
    for (int c = 0; c < 3; ++c) {
      u2[c] = cb[l*3 + c]     - cb[(l-1)*3 + c];
      u1[c] = cb[(l+1)*3 + c] - cb[l*3 + c];
      u0[c] = cb[(l+2)*3 + c] - cb[(l+1)*3 + c];
    }
    norm3(u2[0],u2[1],u2[2]); norm3(u1[0],u1[1],u1[2]); norm3(u0[0],u0[1],u0[2]);
    float n2[3], n1[3];
    cross3(u2, u1, n2); norm3(n2[0],n2[1],n2[2]);
    cross3(u1, u0, n1); norm3(n1[0],n1[1],n1[2]);
    float cosA = -(u1[0]*u0[0] + u1[1]*u0[1] + u1[2]*u0[2]);
    cosA = fminf(fmaxf(cosA, -1.f + 1e-6f), 1.f - 1e-6f);
    float A = acosf(cosA);
    float cosD = n2[0]*n1[0] + n2[1]*n1[1] + n2[2]*n1[2];
    cosD = fminf(fmaxf(cosD, -1.f + 1e-6f), 1.f - 1e-6f);
    float sgn = u2[0]*n1[0] + u2[1]*n1[1] + u2[2]*n1[2];
    float s = (sgn > 0.f) ? 1.f : ((sgn < 0.f) ? -1.f : 0.f);
    float sinA = sinf(A);
    float sinD = sqrtf(fmaxf(0.f, 1.f - cosD*cosD));
    ADv[0] = cosA; ADv[1] = sinA * cosD; ADv[2] = sinA * s * sinD;
    float o1[3] = {u2[0]-u1[0], u2[1]-u1[1], u2[2]-u1[2]};
    norm3(o1[0],o1[1],o1[2]);
    float r1[3]; cross3(o1, n2, r1);
    Ov[0]=o1[0]; Ov[1]=o1[1]; Ov[2]=o1[2];
    Ov[3]=r1[0]; Ov[4]=r1[1]; Ov[5]=r1[2];
    Ov[6]=n2[0]; Ov[7]=n2[1]; Ov[8]=n2[2];
  }
  for (int c = 0; c < 9; ++c) O[n*9 + c]  = Ov[c];
  for (int c = 0; c < 3; ++c) AD[n*3 + c] = ADv[c];
}

__device__ inline void rbf16(unsigned short* out, float D) {
  #pragma unroll
  for (int m = 0; m < 16; ++m) {
    float mu = 2.0f + (20.0f / 15.0f) * (float)m;
    float z = (D - mu) * (1.0f / 1.25f);
    out[m] = f2bfu(expf(-z * z));
  }
}

__global__ void k_edgefeat(const float* __restrict__ ca, const float* __restrict__ mask,
                           const int* __restrict__ ridx, const float* __restrict__ Obuf,
                           const int* __restrict__ eidx, const float* __restrict__ dn,
                           const float* __restrict__ posW, const float* __restrict__ posb,
                           unsigned short* __restrict__ Eraw) {
  int e = blockIdx.x * blockDim.x + threadIdx.x;
  if (e >= NEDGE) return;
  int node = e >> 5;
  int b = node >> 10, l = node & (Ldim - 1);
  int j = eidx[e];
  int nj = (b << 10) + j;
  const float* cb = ca + (long)b * Ldim * 3;
  float Pl[3], Pj[3], P0l[3], P2l[3], P0j[3], P2j[3];
  #pragma unroll
  for (int c = 0; c < 3; ++c) {
    Pl[c]  = cb[l*3 + c];
    Pj[c]  = cb[j*3 + c];
    P0l[c] = (l == 0)        ? 0.f : cb[(l-1)*3 + c];
    P2l[c] = (l == Ldim - 1) ? 0.f : cb[(l+1)*3 + c];
    P0j[c] = (j == 0)        ? 0.f : cb[(j-1)*3 + c];
    P2j[c] = (j == Ldim - 1) ? 0.f : cb[(j+1)*3 + c];
  }
  unsigned short* out = Eraw + (long)e * EPAD;
  int off = ridx[node] - ridx[nj];
  int same = ((int)mask[node] == (int)mask[nj]) ? 1 : 0;
  int dcl = min(max(off + 32, 0), 64);
  int d = dcl * same + (1 - same) * 65;
  #pragma unroll
  for (int c = 0; c < 16; ++c) out[c] = f2bfu(posW[d*16 + c] + posb[c]);
  rbf16(out + 16, dn[e]);
  const float* pa[8] = {P0l, P2l, P0l, P0l, Pl, Pl, P2l, P2l};
  const float* pb[8] = {P0j, P2j, Pj,  P2j, P0j, P2j, P0j, Pj};
  #pragma unroll
  for (int p = 0; p < 8; ++p) {
    float dx = pa[p][0]-pb[p][0], dy = pa[p][1]-pb[p][1], dz = pa[p][2]-pb[p][2];
    float D = sqrtf(dx*dx + dy*dy + dz*dz + 1e-6f);
    rbf16(out + 32 + p*16, D);
  }
  float Om[9], On[9];
  #pragma unroll
  for (int c = 0; c < 9; ++c) { Om[c] = Obuf[(long)node*9 + c]; On[c] = Obuf[(long)nj*9 + c]; }
  float dX[3] = {Pj[0]-Pl[0], Pj[1]-Pl[1], Pj[2]-Pl[2]};
  float u[3];
  #pragma unroll
  for (int i = 0; i < 3; ++i)
    u[i] = Om[i*3+0]*dX[0] + Om[i*3+1]*dX[1] + Om[i*3+2]*dX[2];
  norm3(u[0], u[1], u[2]);
  out[160] = f2bfu(u[0]); out[161] = f2bfu(u[1]); out[162] = f2bfu(u[2]);
  float R[3][3];
  #pragma unroll
  for (int i = 0; i < 3; ++i)
    #pragma unroll
    for (int m = 0; m < 3; ++m)
      R[i][m] = Om[0*3+i]*On[0*3+m] + Om[1*3+i]*On[1*3+m] + Om[2*3+i]*On[2*3+m];
  float Rxx = R[0][0], Ryy = R[1][1], Rzz = R[2][2];
  float mg[3] = {0.5f*sqrtf(fabsf(1.f + Rxx - Ryy - Rzz)),
                 0.5f*sqrtf(fabsf(1.f - Rxx + Ryy - Rzz)),
                 0.5f*sqrtf(fabsf(1.f - Rxx - Ryy + Rzz))};
  float sg[3] = {R[2][1]-R[1][2], R[0][2]-R[2][0], R[1][0]-R[0][1]};
  #pragma unroll
  for (int c = 0; c < 3; ++c) sg[c] = (sg[c] > 0.f) ? 1.f : ((sg[c] < 0.f) ? -1.f : 0.f);
  float w = sqrtf(fmaxf(0.f, 1.f + Rxx + Ryy + Rzz)) * 0.5f;
  float q[4] = {sg[0]*mg[0], sg[1]*mg[1], sg[2]*mg[2], w};
  float qn = sqrtf(q[0]*q[0] + q[1]*q[1] + q[2]*q[2] + q[3]*q[3]);
  float inv = 1.f / fmaxf(qn, 1e-12f);
  #pragma unroll
  for (int c = 0; c < 4; ++c) out[163 + c] = f2bfu(q[c] * inv);
  #pragma unroll
  for (int c = EFDIM; c < EPAD; ++c) out[c] = 0;
}

// ---------------- WMMA tile GEMM (the MI455X matrix path) ----------------
// C[M,N] = act(A[M,K] @ W[K,N] + bias).  A built on the fly per MODE.
// LDS tiles are stored in *fragment order*: each lane's 16-bf16 WMMA operand is
// contiguous (2x ds_load_b128 per operand).
enum { MODE_PLAIN_F32 = 0, MODE_PLAIN_BF16 = 1, MODE_CAT3 = 2, MODE_CAT4 = 3 };

#define FRAG_STRIDE 24   // bf16 per lane slot: 48B -> 16B-aligned, reduced bank conflicts

template<int MODE, int OUT_BF16, int ACT_GELU>
__global__ __launch_bounds__(256)
void k_gemm(const void* __restrict__ P0, const float* __restrict__ P1,
            const float* __restrict__ P2, const float* __restrict__ P3,
            const int* __restrict__ eidx,
            const float* __restrict__ W, const float* __restrict__ bias,
            void* __restrict__ Cout, int Kdim, int Kreal, int Nout, int lda) {
  __shared__ __align__(16) __bf16 Af[8][32][FRAG_STRIDE];  // 8 waves x 16-row A frags
  __shared__ __align__(16) __bf16 Bf[8][32][FRAG_STRIDE];  // 8 n-tiles x 32x16 B frags
  const int tid  = threadIdx.x;
  const int wave = tid >> 5;
  const int lane = tid & 31;
  const int rowbase = blockIdx.x * 128;
  const int n0 = blockIdx.y * 128;

  v8f acc[8];
  #pragma unroll
  for (int t = 0; t < 8; ++t)
    #pragma unroll
    for (int q = 0; q < 8; ++q) acc[t][q] = 0.f;

  const int arow = tid >> 1;        // 0..127 : A tile row handled by this thread
  const int ag   = tid & 1;         // lane-half of the A fragment
  const int bn   = tid >> 1;        // 0..127 : B tile column handled by this thread
  const int bkh  = tid & 1;         // K-half of the B fragment

  for (int k0 = 0; k0 < Kdim; k0 += 32) {
    // ---- stage A (128x32) in fragment order ----
    {
      long r = (long)rowbase + arow;
      const float* srcf = nullptr;
      const unsigned short* srcb = nullptr;
      if (MODE == MODE_PLAIN_F32) {
        srcf = (const float*)P0 + r * lda + k0;
      } else if (MODE == MODE_PLAIN_BF16) {
        srcb = (const unsigned short*)P0 + r * lda + k0;
      } else {
        long e = r;
        int node = (int)(e >> 5);
        int bi = node >> 10;
        if (k0 < 128) {
          srcf = (const float*)P0 + (long)node * 128 + k0;
        } else if (k0 < 256) {
          srcf = P1 + e * 128 + (k0 - 128);
        } else {
          int nbr = (bi << 10) + eidx[e];
          if (MODE == MODE_CAT3) srcf = (const float*)P0 + (long)nbr * 128 + (k0 - 256);
          else srcf = (k0 < 384) ? (P2 + (long)nbr * 128 + (k0 - 256))
                                 : (P3 + (long)nbr * 128 + (k0 - 384));
        }
      }
      // documented 16-bit A 16x32 layout: elems 0..7 -> K=8g..8g+7,
      //                                   elems 8..15 -> K=16+8g..23+8g
      v8bf lo, hi;
      #pragma unroll
      for (int e2 = 0; e2 < 8; ++e2) {
        float v0, v1;
        if (MODE == MODE_PLAIN_BF16) {
          v0 = bfu2f(srcb[8*ag + e2]); v1 = bfu2f(srcb[16 + 8*ag + e2]);
        } else {
          v0 = srcf[8*ag + e2];        v1 = srcf[16 + 8*ag + e2];
        }
        lo[e2] = f2bf(v0); hi[e2] = f2bf(v1);
      }
      v8bf* dst = (v8bf*)&Af[arow >> 4][(ag << 4) | (arow & 15)][0];
      dst[0] = lo; dst[1] = hi;
    }
    // ---- stage B (32x128) in fragment order ----
    {
      const int nt = bn >> 4;
      const int lslot = (bn & 15) | (bkh << 4);
      if (k0 + 32 < Kreal) __builtin_prefetch(&W[(long)(k0 + 32) * Nout + n0 + bn], 0, 1);
      v8bf lo, hi;
      #pragma unroll
      for (int e2 = 0; e2 < 8; ++e2) {
        int gk0 = k0 + bkh * 8 + e2;
        int gk1 = gk0 + 16;
        float w0 = (gk0 < Kreal) ? W[(long)gk0 * Nout + n0 + bn] : 0.f;
        float w1 = (gk1 < Kreal) ? W[(long)gk1 * Nout + n0 + bn] : 0.f;
        lo[e2] = f2bf(w0); hi[e2] = f2bf(w1);
      }
      v8bf* dst = (v8bf*)&Bf[nt][lslot][0];
      dst[0] = lo; dst[1] = hi;
    }
    __syncthreads();

    // ---- WMMA: wave owns 16 rows x (8 x 16) cols ----
    const v8bf* ap = (const v8bf*)&Af[wave][lane][0];
    v8bf a0 = ap[0], a1 = ap[1];
    v16bf afr = __builtin_shufflevector(a0, a1, 0,1,2,3,4,5,6,7,8,9,10,11,12,13,14,15);
    #pragma unroll
    for (int nt = 0; nt < 8; ++nt) {
      const v8bf* bp = (const v8bf*)&Bf[nt][lane][0];
      v8bf b0 = bp[0], b1 = bp[1];
      v16bf bfr = __builtin_shufflevector(b0, b1, 0,1,2,3,4,5,6,7,8,9,10,11,12,13,14,15);
      acc[nt] = __builtin_amdgcn_wmma_f32_16x16x32_bf16(
          false, afr, false, bfr, (short)0, acc[nt], false, false);
    }
    __syncthreads();
  }

  // ---- epilogue: bias + activation + store ----
  const int g = lane >> 4;
  const int cl = lane & 15;
  #pragma unroll
  for (int nt = 0; nt < 8; ++nt) {
    int coln = n0 + nt * 16 + cl;
    float bv = bias ? bias[coln] : 0.f;
    #pragma unroll
    for (int e = 0; e < 8; ++e) {
      long row = (long)rowbase + wave * 16 + e + 8 * g;  // C/D 16x16 f32 layout
      float v = acc[nt][e] + bv;
      if (ACT_GELU) v = gelu_exact(v);
      if (OUT_BF16) ((unsigned short*)Cout)[row * Nout + coln] = f2bfu(v);
      else          ((float*)Cout)[row * Nout + coln] = v;
    }
  }
}

// ---------------- small fused kernels ----------------
__global__ void k_node_emb(const float* __restrict__ AD, const float* __restrict__ W,
                           const float* __restrict__ g, const float* __restrict__ b,
                           float* __restrict__ V) {
  __shared__ float s1[128], s2[128];
  int n = blockIdx.x, h = threadIdx.x;
  float x = AD[n*3]*W[h] + AD[n*3+1]*W[128+h] + AD[n*3+2]*W[256+h];
  V[(long)n*128 + h] = block_ln128(x, h, g, b, s1, s2);
}

__global__ void k_ln_bf16(unsigned short* __restrict__ X, const float* __restrict__ g,
                          const float* __restrict__ b) {
  __shared__ float s1[128], s2[128];
  long r = blockIdx.x; int h = threadIdx.x;
  float x = bfu2f(X[r*128 + h]);
  X[r*128 + h] = f2bfu(block_ln128(x, h, g, b, s1, s2));
}

__global__ void k_add_ln(const float* __restrict__ resid, const unsigned short* __restrict__ add,
                         const float* __restrict__ g, const float* __restrict__ b,
                         float* __restrict__ out, const float* __restrict__ mask) {
  __shared__ float s1[128], s2[128];
  long r = blockIdx.x; int h = threadIdx.x;
  float x = resid[r*128 + h] + bfu2f(add[r*128 + h]);
  float y = block_ln128(x, h, g, b, s1, s2);
  if (mask) y *= mask[r];
  out[r*128 + h] = y;
}

__global__ void k_reduce_ln(const unsigned short* __restrict__ m, const int* __restrict__ eidx,
                            const float* __restrict__ mask, const float* __restrict__ resid,
                            const float* __restrict__ g, const float* __restrict__ b,
                            float* __restrict__ out, int use_mask) {
  __shared__ float s1[128], s2[128];
  int n = blockIdx.x, h = threadIdx.x;
  int bidx = n >> 10;
  const unsigned short* mp = m + (long)n * Kneib * 128 + h;
  float acc = 0.f;
  for (int k = 0; k < Kneib; ++k) {
    float v = bfu2f(mp[k * 128]);
    if (use_mask) v *= mask[n] * mask[(bidx << 10) + eidx[n*Kneib + k]];
    acc += v;
  }
  float x = resid[(long)n*128 + h] + acc * (1.0f / 30.0f);
  out[(long)n*128 + h] = block_ln128(x, h, g, b, s1, s2);
}

__global__ void k_film_ss(const float* __restrict__ t, const float* __restrict__ tW,
                          const float* __restrict__ tb, float* __restrict__ ss) {
  __shared__ float st[TCDdim];
  int b = blockIdx.x, tid = threadIdx.x;
  for (int i = tid; i < TCDdim; i += 256) {
    float v = t[b*TCDdim + i];
    st[i] = v / (1.0f + expf(-v));         // silu
  }
  __syncthreads();
  float acc = tb[tid];
  for (int j = 0; j < TCDdim; ++j) acc += st[j] * tW[j*256 + tid];
  ss[b*256 + tid] = acc;
}

__global__ void k_film_apply(const float* __restrict__ hV, const float* __restrict__ ss,
                             float* __restrict__ outV) {
  int n = blockIdx.x, h = threadIdx.x;
  int b = n >> 10;
  float s0 = ss[b*256 + h], s1 = ss[b*256 + 128 + h];
  outV[(long)n*128 + h] = hV[(long)n*128 + h] * (s0 + 1.0f) + s1;
}

__global__ void k_hs_lookup(const int* __restrict__ aat, const float* __restrict__ Ws,
                            float* __restrict__ hS) {
  int n = blockIdx.x, h = threadIdx.x;
  hS[(long)n*128 + h] = Ws[(long)aat[n]*128 + h];
}

// ---------------- host orchestration ----------------
extern "C" void kernel_launch(void* const* d_in, const int* in_sizes, int n_in,
                              void* d_out, int out_size, void* d_ws, size_t ws_size,
                              hipStream_t stream) {
  const float* coords = (const float*)d_in[0];
  const int*   aatype = (const int*)  d_in[1];
  const float* mask   = (const float*)d_in[2];
  const int*   ridx   = (const int*)  d_in[3];
  const float* tcond  = (const float*)d_in[4];
  const float* pos_W = (const float*)d_in[5];
  const float* pos_b = (const float*)d_in[6];
  const float* node_emb_W = (const float*)d_in[7];
  const float* norm_n_g = (const float*)d_in[8];
  const float* norm_n_b = (const float*)d_in[9];
  const float* edge_emb_W = (const float*)d_in[10];
  const float* norm_e_g = (const float*)d_in[11];
  const float* norm_e_b = (const float*)d_in[12];
  const float* Wv_W = (const float*)d_in[13];
  const float* Wv_b = (const float*)d_in[14];
  const float* We_W = (const float*)d_in[15];
  const float* We_b = (const float*)d_in[16];
  const float* Ws   = (const float*)d_in[17];
  const float* enc_W1  = (const float*)d_in[18]; const float* enc_b1  = (const float*)d_in[19];
  const float* enc_W2  = (const float*)d_in[20]; const float* enc_b2  = (const float*)d_in[21];
  const float* enc_W3  = (const float*)d_in[22]; const float* enc_b3  = (const float*)d_in[23];
  const float* enc_n1g = (const float*)d_in[24]; const float* enc_n1b = (const float*)d_in[25];
  const float* enc_n2g = (const float*)d_in[26]; const float* enc_n2b = (const float*)d_in[27];
  const float* enc_Win = (const float*)d_in[28]; const float* enc_bin = (const float*)d_in[29];
  const float* enc_Wout= (const float*)d_in[30]; const float* enc_bout= (const float*)d_in[31];
  const float* enc_tW  = (const float*)d_in[32]; const float* enc_tb  = (const float*)d_in[33];
  const float* dec_W1  = (const float*)d_in[34]; const float* dec_b1  = (const float*)d_in[35];
  const float* dec_W2  = (const float*)d_in[36]; const float* dec_b2  = (const float*)d_in[37];
  const float* dec_W3  = (const float*)d_in[38]; const float* dec_b3  = (const float*)d_in[39];
  const float* dec_n1g = (const float*)d_in[40]; const float* dec_n1b = (const float*)d_in[41];
  const float* dec_n2g = (const float*)d_in[42]; const float* dec_n2b = (const float*)d_in[43];
  const float* dec_Win = (const float*)d_in[44]; const float* dec_bin = (const float*)d_in[45];
  const float* dec_Wout= (const float*)d_in[46]; const float* dec_bout= (const float*)d_in[47];
  const float* dec_tW  = (const float*)d_in[48]; const float* dec_tb  = (const float*)d_in[49];
  const float* enc_We1 = (const float*)d_in[50]; const float* enc_be1 = (const float*)d_in[51];
  const float* enc_We2 = (const float*)d_in[52]; const float* enc_be2 = (const float*)d_in[53];
  const float* enc_We3 = (const float*)d_in[54]; const float* enc_be3 = (const float*)d_in[55];
  const float* enc_n3g = (const float*)d_in[56]; const float* enc_n3b = (const float*)d_in[57];

  char* wp = (char*)d_ws;
  auto alloc = [&](size_t bytes) -> void* {
    void* p = (void*)wp;
    wp += (bytes + 255) & ~(size_t)255;
    return p;
  };
  float* ca   = (float*)alloc((size_t)NNODE * 3 * 4);
  int*   eidx = (int*)  alloc((size_t)NEDGE * 4);
  float* dn   = (float*)alloc((size_t)NEDGE * 4);
  float* Obuf = (float*)alloc((size_t)NNODE * 9 * 4);
  float* ADb  = (float*)alloc((size_t)NNODE * 3 * 4);
  float* hV   = (float*)alloc((size_t)NNODE * 128 * 4);
  float* hVf  = (float*)alloc((size_t)NNODE * 128 * 4);
  float* hS   = (float*)alloc((size_t)NNODE * 128 * 4);
  float* ssb  = (float*)alloc((size_t)Bdim * 256 * 4);
  unsigned short* ffh = (unsigned short*)alloc((size_t)NNODE * 512 * 2);
  unsigned short* ff2 = (unsigned short*)alloc((size_t)NNODE * 128 * 2);
  float* hE = (float*)alloc((size_t)NEDGE * 128 * 4);
  unsigned short* m1 = (unsigned short*)alloc((size_t)NEDGE * EPAD * 2); // Eraw region, reused
  unsigned short* m2 = (unsigned short*)alloc((size_t)NEDGE * 128 * 2);

  const dim3 blk256(256), blk128(128), blk64(64);
  const dim3 gE((NEDGE) / 128, 1);
  const dim3 gN((NNODE) / 128, 1);
  const dim3 gN4((NNODE) / 128, 4);

  // ---- features ----
  k_extract_ca<<<(NNODE*3 + 255)/256, blk256, 0, stream>>>(coords, ca);
  k_knn<<<(NNODE + 63)/64, blk64, 0, stream>>>(ca, mask, eidx, dn);
  k_orient<<<(NNODE + 127)/128, blk128, 0, stream>>>(ca, Obuf, ADb);
  k_edgefeat<<<(NEDGE + 127)/128, blk128, 0, stream>>>(ca, mask, ridx, Obuf, eidx, dn,
                                                       pos_W, pos_b, m1 /*Eraw*/);
  k_node_emb<<<NNODE, blk128, 0, stream>>>(ADb, node_emb_W, norm_n_g, norm_n_b, hVf);
  k_gemm<MODE_PLAIN_F32, 0, 0><<<gN, blk256, 0, stream>>>(hVf, nullptr, nullptr, nullptr,
      nullptr, Wv_W, Wv_b, hV, 128, 128, 128, 128);
  k_gemm<MODE_PLAIN_BF16, 1, 0><<<gE, blk256, 0, stream>>>(m1, nullptr, nullptr, nullptr,
      nullptr, edge_emb_W, nullptr, m2, EPAD, EFDIM, 128, EPAD);
  k_ln_bf16<<<NEDGE, blk128, 0, stream>>>(m2, norm_e_g, norm_e_b);
  k_gemm<MODE_PLAIN_BF16, 0, 0><<<gE, blk256, 0, stream>>>(m2, nullptr, nullptr, nullptr,
      nullptr, We_W, We_b, hE, 128, 128, 128, 128);

  // ---- encoder ----
  for (int i = 0; i < 3; ++i) {
    k_film_ss<<<Bdim, blk256, 0, stream>>>(tcond, enc_tW + (size_t)i*512*256, enc_tb + i*256, ssb);
    k_film_apply<<<NNODE, blk128, 0, stream>>>(hV, ssb, hVf);
    k_gemm<MODE_CAT3, 1, 1><<<gE, blk256, 0, stream>>>(hVf, hE, nullptr, nullptr, eidx,
        enc_W1 + (size_t)i*384*128, enc_b1 + i*128, m1, 384, 384, 128, 0);
    k_gemm<MODE_PLAIN_BF16, 1, 1><<<gE, blk256, 0, stream>>>(m1, nullptr, nullptr, nullptr,
        nullptr, enc_W2 + (size_t)i*128*128, enc_b2 + i*128, m2, 128, 128, 128, 128);
    k_gemm<MODE_PLAIN_BF16, 1, 0><<<gE, blk256, 0, stream>>>(m2, nullptr, nullptr, nullptr,
        nullptr, enc_W3 + (size_t)i*128*128, enc_b3 + i*128, m1, 128, 128, 128, 128);
    k_reduce_ln<<<NNODE, blk128, 0, stream>>>(m1, eidx, mask, hVf,
        enc_n1g + i*128, enc_n1b + i*128, hV, 1);
    k_gemm<MODE_PLAIN_F32, 1, 1><<<gN4, blk256, 0, stream>>>(hV, nullptr, nullptr, nullptr,
        nullptr, enc_Win + (size_t)i*128*512, enc_bin + i*512, ffh, 128, 128, 512, 128);
    k_gemm<MODE_PLAIN_BF16, 1, 0><<<gN, blk256, 0, stream>>>(ffh, nullptr, nullptr, nullptr,
        nullptr, enc_Wout + (size_t)i*512*128, enc_bout + i*128, ff2, 512, 512, 128, 512);
    k_add_ln<<<NNODE, blk128, 0, stream>>>(hV, ff2, enc_n2g + i*128, enc_n2b + i*128, hV, mask);
    k_gemm<MODE_CAT3, 1, 1><<<gE, blk256, 0, stream>>>(hV, hE, nullptr, nullptr, eidx,
        enc_We1 + (size_t)i*384*128, enc_be1 + i*128, m1, 384, 384, 128, 0);
    k_gemm<MODE_PLAIN_BF16, 1, 1><<<gE, blk256, 0, stream>>>(m1, nullptr, nullptr, nullptr,
        nullptr, enc_We2 + (size_t)i*128*128, enc_be2 + i*128, m2, 128, 128, 128, 128);
    k_gemm<MODE_PLAIN_BF16, 1, 0><<<gE, blk256, 0, stream>>>(m2, nullptr, nullptr, nullptr,
        nullptr, enc_We3 + (size_t)i*128*128, enc_be3 + i*128, m1, 128, 128, 128, 128);
    k_add_ln<<<NEDGE, blk128, 0, stream>>>(hE, m1, enc_n3g + i*128, enc_n3b + i*128, hE, nullptr);
  }
  hipMemcpyAsync((float*)d_out + (size_t)NNODE*128, hV, (size_t)NNODE*128*4,
                 hipMemcpyDeviceToDevice, stream);

  // ---- decoder ----
  k_hs_lookup<<<NNODE, blk128, 0, stream>>>(aatype, Ws, hS);
  for (int i = 0; i < 3; ++i) {
    k_film_ss<<<Bdim, blk256, 0, stream>>>(tcond, dec_tW + (size_t)i*512*256, dec_tb + i*256, ssb);
    k_film_apply<<<NNODE, blk128, 0, stream>>>(hV, ssb, hVf);
    k_gemm<MODE_CAT4, 1, 1><<<gE, blk256, 0, stream>>>(hVf, hE, hS, hV, eidx,
        dec_W1 + (size_t)i*512*128, dec_b1 + i*128, m1, 512, 512, 128, 0);
    k_gemm<MODE_PLAIN_BF16, 1, 1><<<gE, blk256, 0, stream>>>(m1, nullptr, nullptr, nullptr,
        nullptr, dec_W2 + (size_t)i*128*128, dec_b2 + i*128, m2, 128, 128, 128, 128);
    k_gemm<MODE_PLAIN_BF16, 1, 0><<<gE, blk256, 0, stream>>>(m2, nullptr, nullptr, nullptr,
        nullptr, dec_W3 + (size_t)i*128*128, dec_b3 + i*128, m1, 128, 128, 128, 128);
    k_reduce_ln<<<NNODE, blk128, 0, stream>>>(m1, eidx, mask, hVf,
        dec_n1g + i*128, dec_n1b + i*128, hV, 0);
    k_gemm<MODE_PLAIN_F32, 1, 1><<<gN4, blk256, 0, stream>>>(hV, nullptr, nullptr, nullptr,
        nullptr, dec_Win + (size_t)i*128*512, dec_bin + i*512, ffh, 128, 128, 512, 128);
    k_gemm<MODE_PLAIN_BF16, 1, 0><<<gN, blk256, 0, stream>>>(ffh, nullptr, nullptr, nullptr,
        nullptr, dec_Wout + (size_t)i*512*128, dec_bout + i*128, ff2, 512, 512, 128, 512);
    k_add_ln<<<NNODE, blk128, 0, stream>>>(hV, ff2, dec_n2g + i*128, dec_n2b + i*128, hV, mask);
  }
  hipMemcpyAsync(d_out, hV, (size_t)NNODE*128*4, hipMemcpyDeviceToDevice, stream);
}
